// NeighborhoodPool_2001454760243
// MI455X (gfx1250) — matrix operand order repaired and checked by reading, not compiled, not run on hardware
//
#include <hip/hip_runtime.h>
#include <hip/hip_bf16.h>

typedef __attribute__((ext_vector_type(16))) _Float16 v16h;
typedef __attribute__((ext_vector_type(8)))  float    v8f;

#define NN   8192
#define FF   256
#define CC   32
#define E0   262144      /* N*DEG original edges */
#define EE   270336      /* E0 + N self loops    */
#define SENT 67108864u   /* N*N                  */

/* d_out flat layout (floats) */
#define OUTX 0
#define OUTE 2097152
#define OUTP 2621440
#define OUTS 2646016
#define OUTN 2654208

__device__ __forceinline__ float selu_f(float v) {
  const float a = 1.6732632423543772f, s = 1.0507009873554805f;
  return v > 0.f ? s * v : s * a * (expf(v) - 1.f);
}
__device__ __forceinline__ unsigned ordf(float f) {
  unsigned b = __float_as_uint(f);
  return (b & 0x80000000u) ? ~b : (b | 0x80000000u);
}
__device__ __forceinline__ float unordf(unsigned u) {
  return (u & 0x80000000u) ? __uint_as_float(u & 0x7fffffffu) : __uint_as_float(~u);
}

/* ---------- init: zero accumulators, prefill edge_pool with -1 ---------- */
__global__ void init_kernel(float* agg1, float* sacc, unsigned* xpu, float* out) {
  int i = blockIdx.x * blockDim.x + threadIdx.x;
  int stride = gridDim.x * blockDim.x;
  for (int k = i; k < NN * CC; k += stride) agg1[k] = 0.f;
  for (int k = i; k < NN; k += stride) sacc[k] = 0.f;
  for (int k = i; k < NN * FF; k += stride) xpu[k] = 0u;
  for (int k = i; k < 2 * E0; k += stride) out[OUTE + k] = -1.0f;
}

/* ---------- build src/dst (+self loops) and spherical features f3 ---------- */
__global__ void build_kernel(const int* ei, const float* pos,
                             int* src, int* dst, float* f3) {
  int e = blockIdx.x * blockDim.x + threadIdx.x;
  if (e >= EE) return;
  int s, d;
  if (e < E0) { s = ei[e]; d = ei[E0 + e]; } else { s = e - E0; d = s; }
  src[e] = s; dst[e] = d;
  float x = pos[d * 3 + 0] - pos[s * 3 + 0];
  float y = pos[d * 3 + 1] - pos[s * 3 + 1];
  float z = pos[d * 3 + 2] - pos[s * 3 + 2];
  float sq = x * x + y * y + z * z;
  bool zero = (sq == 0.f);
  float rho = zero ? 0.f : sqrtf(sq);
  bool m0 = (x == 0.f) && (y == 0.f);
  float theta = (m0 || zero) ? 0.f : atan2f(y, x);
  float phi = zero ? 0.f : asinf(fminf(1.f, fmaxf(-1.f, z / rho)));
  const float invpi = 0.31830988618379067f;
  f3[e * 3 + 0] = rho;
  f3[e * 3 + 1] = theta * invpi;
  f3[e * 3 + 2] = phi * invpi;
}

/* ---------- pre-swizzle g2_W2 into per-lane WMMA B fragments (f16) ----------
   w2b[((t*32 + lane)*16) + k] = W2[row][col], col = t*16 + (lane&15),
   row = (lane>>4)*8 + (k&7) + (k>>3)*16  -> each lane's 16-half fragment is contiguous,
   32-byte aligned, loadable as a single v16h (two global_load_b128).      */
__global__ void w2b_kernel(const float* W2, _Float16* w2b) {
  int i = blockIdx.x * blockDim.x + threadIdx.x;
  if (i >= 2 * 32 * 16) return;
  int t = i >> 9, lane = (i >> 4) & 31, k = i & 15;
  int col = t * 16 + (lane & 15);
  int row = (lane >> 4) * 8 + (k & 7) + (k >> 3) * 16;
  w2b[i] = (_Float16)W2[row * CC + col];
}

/* ---------- GKC layer 1 (x=pos, cout=32): lane = channel ---------- */
__global__ void gkc1_kernel(const int* src, const int* dst, const float* f3,
                            const float* pos, const float* W1, const float* b1,
                            const float* W2, const float* b2, float* agg1) {
  __shared__ float Bs[8][CC], Vs[8][CC], XJ[8][4], WCH[CC];
  int tid = threadIdx.x, g = tid >> 5, lane = tid & 31;
  if (tid < CC) WCH[tid] = W1[3 * CC + tid];
  __syncthreads();
  int e = blockIdx.x * 8 + g;
  if (e >= EE) return;
  int s = src[e];
  if (lane < 3) XJ[g][lane] = pos[s * 3 + lane];
  float f0 = f3[e * 3 + 0], f1 = f3[e * 3 + 1], f2 = f3[e * 3 + 2];
  int j = lane;  /* lane j computes hidden component j for this edge */
  Bs[g][j] = f0 * W1[j] + f1 * W1[CC + j] + f2 * W1[2 * CC + j] + b1[j];
  Vs[g][j] = W2[j * 3 + 0] * XJ[g][0] + W2[j * 3 + 1] * XJ[g][1] + W2[j * 3 + 2] * XJ[g][2];
  float b2x = b2[0] * XJ[g][0] + b2[1] * XJ[g][1] + b2[2] * XJ[g][2];
  float c = (float)lane, acc = 0.f;  /* lane = output channel */
#pragma unroll 8
  for (int jj = 0; jj < CC; ++jj)
    acc += selu_f(Bs[g][jj] + c * WCH[jj]) * Vs[g][jj];
  atomicAdd(&agg1[dst[e] * CC + lane], acc + b2x);
}

/* ---------- h1 = selu(agg1 + g1_bias) ---------- */
__global__ void h1fin_kernel(const float* agg1, const float* bias, float* h1) {
  int i = blockIdx.x * blockDim.x + threadIdx.x;
  if (i < NN * CC) h1[i] = selu_f(agg1[i] + bias[i & (CC - 1)]);
}

/* ---------- GKC layer 2 via WMMA: per wave a 16-edge tile, K = H @ W2 ---------- */
__global__ void gkc2_kernel(const int* src, const int* dst, const float* f3,
                            const float* h1, const float* W1, const float* b1,
                            const _Float16* w2b, const float* b2, float* sacc) {
  __shared__ _Float16 Hs[8][16][CC];   /* selu(base2) tile, f16           */
  __shared__ float    Ks[8][16][CC];   /* H @ W2 result tile              */
  __shared__ float    F3s[8][16][3];
  int tid = threadIdx.x, w = tid >> 5, lane = tid & 31;
  int m = lane & 15, hi = lane >> 4;
  int tileBase = (blockIdx.x * 8 + w) * 16;

  if (lane < 16) {
    int ge = tileBase + lane;
#pragma unroll
    for (int k = 0; k < 3; ++k) F3s[w][lane][k] = (ge < EE) ? f3[ge * 3 + k] : 0.f;
  }
  /* lane j fills hidden column j for all 16 edges (same-wave LDS, in-order) */
  {
    int j = lane;
    float w0 = W1[j], w1 = W1[CC + j], w2 = W1[2 * CC + j], bb = b1[j];
#pragma unroll
    for (int e = 0; e < 16; ++e) {
      float base = F3s[w][e][0] * w0 + F3s[w][e][1] * w1 + F3s[w][e][2] * w2 + bb;
      Hs[w][e][j] = (_Float16)selu_f(base);
    }
  }
  /* A fragment: 16x32 f16. lane<16: M=lane, K={0..7,16..23};
     lane>=16: M=lane-16, K={8..15,24..31}  (two contiguous b128 LDS loads) */
  v16h a;
#pragma unroll
  for (int i = 0; i < 8; ++i) {
    a[i]     = Hs[w][m][hi * 8 + i];
    a[8 + i] = Hs[w][m][16 + hi * 8 + i];
  }
  /* two 16-column output tiles; B fragment is one aligned v16h per lane */
#pragma unroll
  for (int t = 0; t < 2; ++t) {
    v16h b = *(const v16h*)(w2b + ((t * 32 + lane) << 4));
    v8f c = {};
    v8f d = __builtin_amdgcn_wmma_f32_16x16x32_f16(false, a, false, b,
                                                   (short)0, c, false, false);
#pragma unroll
    for (int r = 0; r < 8; ++r) Ks[w][r + hi * 8][t * 16 + m] = d[r];
  }
  /* per-edge dot: lane handles edge m, half hi of the 32 features */
  int ge = tileBase + m;
  if (ge < EE) {
    int s = src[ge];
    float sum = 0.f;
#pragma unroll
    for (int j = hi * 16; j < hi * 16 + 16; ++j)
      sum += (Ks[w][m][j] + b2[j]) * h1[s * CC + j];
    atomicAdd(&sacc[dst[ge]], sum);
  }
}

/* ---------- score = sacc + g2_bias; pack stable sort keys ---------- */
__global__ void scorefin_kernel(const float* sacc, const float* bias,
                                float* out, unsigned long long* keys) {
  int i = blockIdx.x * blockDim.x + threadIdx.x;
  if (i >= NN) return;
  float s = sacc[i] + bias[0];
  out[OUTS + i] = s;
  keys[i] = ((unsigned long long)ordf(s) << 32) | (unsigned)i;
}

/* ---------- bitonic compare-exchange passes ---------- */
__global__ void bitonic64_kernel(unsigned long long* a, int j, int k) {
  int i = blockIdx.x * blockDim.x + threadIdx.x;
  int ixj = i ^ j;
  if (ixj > i) {
    unsigned long long ai = a[i], aj = a[ixj];
    bool up = ((i & k) == 0);
    if ((ai > aj) == up) { a[i] = aj; a[ixj] = ai; }
  }
}
__global__ void bitonic32_kernel(unsigned* a, int j, int k) {
  int i = blockIdx.x * blockDim.x + threadIdx.x;
  int ixj = i ^ j;
  if (ixj > i) {
    unsigned ai = a[i], aj = a[ixj];
    bool up = ((i & k) == 0);
    if ((ai > aj) == up) { a[i] = aj; a[ixj] = ai; }
  }
}

/* ---------- serial greedy clustering: one workgroup, masks in LDS ---------- */
__global__ __launch_bounds__(1024) void cluster_kernel(
    const unsigned long long* keys, const int* src, const int* dst,
    int* ealive, int* cluster, int* nodes, int* ncnt, float* out) {
  __shared__ int nm0s[NN];
  __shared__ unsigned char nm1s[NN];
  __shared__ int red[1024];
  int tid = threadIdx.x;
  for (int i = tid; i < NN; i += 1024) { nm0s[i] = 1; cluster[i] = 0; }
  for (int e = tid; e < EE; e += 1024) ealive[e] = 1;  /* thread-owned slices */
  __syncthreads();
  int n = 0;
  for (int iter = 0; iter <= NN; ++iter) {
    /* first alive node in ascending score order */
    int best = 0x7fffffff;
    for (int p = tid; p < NN; p += 1024) {
      int nd = (int)(keys[p] & 0xffffffffu);
      if (nm0s[nd]) { best = p; break; }
    }
    red[tid] = best;
    __syncthreads();
    for (int s = 512; s > 0; s >>= 1) {
      if (tid < s) red[tid] = min(red[tid], red[tid + s]);
      __syncthreads();
    }
    int bp = red[0];
    if (bp == 0x7fffffff) break;
    int node = (int)(keys[bp] & 0xffffffffu);
    for (int i = tid; i < NN; i += 1024) nm1s[i] = 0;
    __syncthreads();
    for (int e = tid; e < EE; e += 1024)
      if (ealive[e] && src[e] == node) nm1s[dst[e]] = 1;
    __syncthreads();
    for (int i = tid; i < NN; i += 1024)
      if (nm1s[i]) { cluster[i] = n; nm0s[i] = 0; }
    if (tid == 0) nodes[n] = node;
    __syncthreads();
    for (int e = tid; e < EE; e += 1024)
      if (ealive[e] && (nm1s[src[e]] || nm1s[dst[e]])) ealive[e] = 0;
    __syncthreads();
    ++n;
  }
  if (tid == 0) { *ncnt = n; out[OUTN] = (float)n; }
}

/* ---------- x_pool: segment-max via monotonic uint atomics ---------- */
__global__ void xscatter_kernel(const float* x, const int* cluster, unsigned* xpu) {
  int idx = blockIdx.x * blockDim.x + threadIdx.x;
  if (idx >= NN * FF) return;
  int v = idx >> 8, f = idx & (FF - 1);
  atomicMax(&xpu[cluster[v] * FF + f], ordf(x[idx]));
}
__global__ void xdecode_kernel(const unsigned* xpu, const int* ncnt, float* out) {
  int idx = blockIdx.x * blockDim.x + threadIdx.x;
  if (idx >= NN * FF) return;
  int i = idx >> 8;
  out[OUTX + idx] = (i < *ncnt) ? unordf(xpu[idx]) : 0.f;
}

/* ---------- pos_pool ---------- */
__global__ void pospool_kernel(const float* pos, const int* nodes,
                               const int* ncnt, float* out) {
  int idx = blockIdx.x * blockDim.x + threadIdx.x;
  if (idx >= NN * 3) return;
  int i = idx / 3;
  out[OUTP + idx] = (i < *ncnt) ? pos[nodes[i] * 3 + (idx % 3)] : 0.f;
}

/* ---------- edge_pool codes ---------- */
__global__ void codes_kernel(const int* ei, const int* cluster, unsigned* codes) {
  int e = blockIdx.x * blockDim.x + threadIdx.x;
  if (e >= E0) return;
  int cs = cluster[ei[e]], cd = cluster[ei[E0 + e]];
  codes[e] = (cs != cd) ? (unsigned)(cs * NN + cd) : SENT;
}

/* ---------- unique-compaction prefix sum (3 kernels) ---------- */
__global__ void p1_kernel(const unsigned* codes, int* bsums) {
  __shared__ int sh[256];
  int i = blockIdx.x * 256 + threadIdx.x;
  unsigned c = codes[i];
  int keep = (c < SENT) && (i == 0 || codes[i - 1] != c);
  sh[threadIdx.x] = keep;
  __syncthreads();
  for (int s = 128; s > 0; s >>= 1) {
    if (threadIdx.x < s) sh[threadIdx.x] += sh[threadIdx.x + s];
    __syncthreads();
  }
  if (threadIdx.x == 0) bsums[blockIdx.x] = sh[0];
}
__global__ void p2_kernel(int* bsums) {
  __shared__ int sh[1024];
  int t = threadIdx.x;
  int orig = bsums[t];
  sh[t] = orig;
  __syncthreads();
  for (int off = 1; off < 1024; off <<= 1) {
    int v = (t >= off) ? sh[t - off] : 0;
    __syncthreads();
    sh[t] += v;
    __syncthreads();
  }
  bsums[t] = sh[t] - orig;  /* exclusive */
}
__global__ void p3_kernel(const unsigned* codes, const int* bofs, float* out) {
  __shared__ int sh[256];
  int i = blockIdx.x * 256 + threadIdx.x;
  unsigned c = codes[i];
  int keep = (c < SENT) && (i == 0 || codes[i - 1] != c);
  sh[threadIdx.x] = keep;
  __syncthreads();
  for (int off = 1; off < 256; off <<= 1) {
    int v = (threadIdx.x >= off) ? sh[threadIdx.x - off] : 0;
    __syncthreads();
    sh[threadIdx.x] += v;
    __syncthreads();
  }
  int pos = bofs[blockIdx.x] + sh[threadIdx.x] - keep;
  if (keep) {
    out[OUTE + pos]      = (float)(c / (unsigned)NN);
    out[OUTE + E0 + pos] = (float)(c % (unsigned)NN);
  }
}

extern "C" void kernel_launch(void* const* d_in, const int* in_sizes, int n_in,
                              void* d_out, int out_size, void* d_ws, size_t ws_size,
                              hipStream_t stream) {
  const float* x     = (const float*)d_in[0];
  const int*   ei    = (const int*)d_in[1];
  const float* pos   = (const float*)d_in[2];
  const float* g1W1  = (const float*)d_in[3];
  const float* g1b1  = (const float*)d_in[4];
  const float* g1W2  = (const float*)d_in[5];
  const float* g1b2  = (const float*)d_in[6];
  const float* g1bi  = (const float*)d_in[7];
  const float* g2W1  = (const float*)d_in[8];
  const float* g2b1  = (const float*)d_in[9];
  const float* g2W2  = (const float*)d_in[10];
  const float* g2b2  = (const float*)d_in[11];
  const float* g2bi  = (const float*)d_in[12];
  float* out = (float*)d_out;

  /* workspace layout */
  char* base = (char*)d_ws;
  size_t o = 0;
  auto alloc = [&](size_t bytes) { char* p = base + o; o = (o + bytes + 255) & ~(size_t)255; return p; };
  int*      src    = (int*)alloc(EE * 4);
  int*      dst    = (int*)alloc(EE * 4);
  float*    f3     = (float*)alloc((size_t)EE * 3 * 4);
  float*    agg1   = (float*)alloc((size_t)NN * CC * 4);
  float*    h1     = (float*)alloc((size_t)NN * CC * 4);
  _Float16* w2b    = (_Float16*)alloc(2 * 32 * 16 * 2);
  float*    sacc   = (float*)alloc(NN * 4);
  unsigned long long* keys = (unsigned long long*)alloc(NN * 8);
  int*      ealive = (int*)alloc(EE * 4);
  int*      clus   = (int*)alloc(NN * 4);
  int*      nodes  = (int*)alloc(NN * 4);
  int*      ncnt   = (int*)alloc(16);
  unsigned* xpu    = (unsigned*)alloc((size_t)NN * FF * 4);
  unsigned* codes  = (unsigned*)alloc(E0 * 4);
  int*      bsums  = (int*)alloc(1024 * 4);
  (void)ws_size; (void)in_sizes; (void)n_in; (void)out_size; (void)x;

  init_kernel<<<2048, 256, 0, stream>>>(agg1, sacc, xpu, out);
  build_kernel<<<(EE + 255) / 256, 256, 0, stream>>>(ei, pos, src, dst, f3);
  w2b_kernel<<<4, 256, 0, stream>>>(g2W2, w2b);

  gkc1_kernel<<<(EE + 7) / 8, 256, 0, stream>>>(src, dst, f3, pos,
                                                g1W1, g1b1, g1W2, g1b2, agg1);
  h1fin_kernel<<<(NN * CC + 255) / 256, 256, 0, stream>>>(agg1, g1bi, h1);

  gkc2_kernel<<<(EE + 127) / 128, 256, 0, stream>>>(src, dst, f3, h1,
                                                    g2W1, g2b1, w2b, g2b2, sacc);
  scorefin_kernel<<<NN / 256, 256, 0, stream>>>(sacc, g2bi, out, keys);

  for (int k = 2; k <= NN; k <<= 1)
    for (int j = k >> 1; j > 0; j >>= 1)
      bitonic64_kernel<<<NN / 256, 256, 0, stream>>>(keys, j, k);

  cluster_kernel<<<1, 1024, 0, stream>>>(keys, src, dst, ealive, clus, nodes, ncnt, out);

  xscatter_kernel<<<(NN * FF + 255) / 256, 256, 0, stream>>>(x, clus, xpu);
  xdecode_kernel<<<(NN * FF + 255) / 256, 256, 0, stream>>>(xpu, ncnt, out);
  pospool_kernel<<<(NN * 3 + 255) / 256, 256, 0, stream>>>(pos, nodes, ncnt, out);

  codes_kernel<<<E0 / 256, 256, 0, stream>>>(ei, clus, codes);
  for (int k = 2; k <= E0; k <<= 1)
    for (int j = k >> 1; j > 0; j >>= 1)
      bitonic32_kernel<<<E0 / 256, 256, 0, stream>>>(codes, j, k);

  p1_kernel<<<E0 / 256, 256, 0, stream>>>(codes, bsums);
  p2_kernel<<<1, 1024, 0, stream>>>(bsums);
  p3_kernel<<<E0 / 256, 256, 0, stream>>>(codes, bsums, out);
}
